// DeformableConvV2_40553081208980
// MI455X (gfx1250) — matrix operand order repaired and checked by reading, not compiled, not run on hardware
//
#include <hip/hip_runtime.h>
#include <hip/hip_bf16.h>
#include <stdint.h>

// ---------------- CDNA5 WMMA types ----------------
typedef __attribute__((ext_vector_type(16))) __bf16 v16bf;
typedef __attribute__((ext_vector_type(8)))  float  v8f;

#define BSZ   8
#define CIN   64
#define COUT  64
#define HD    128
#define WD    128
#define HW    (HD*WD)      // 16384
#define KTOT  576          // 9 taps * 64 channels, k = tap*64 + c
#define NCHUNK 18          // 576 / 32

union FragBF { uint32_t u[8]; v16bf v; };

__device__ __forceinline__ uint32_t pack2bf(float a, float b) {
    union { __bf16 h[2]; uint32_t u; } t;
    t.h[0] = (__bf16)a; t.h[1] = (__bf16)b;
    return t.u;
}
__device__ __forceinline__ uint16_t f2bf_bits(float f) {
    union { __bf16 h; uint16_t s; } t; t.h = (__bf16)f; return t.s;
}
__device__ __forceinline__ float2 bf2x(uint32_t u) {
    union { uint32_t u; __bf16 h[2]; } t; t.u = u;
    return make_float2((float)t.h[0], (float)t.h[1]);
}
__device__ __forceinline__ uint32_t lerp_pack(uint32_t a00, uint32_t a01,
                                              uint32_t a10, uint32_t a11,
                                              float w00, float w01, float w10, float w11) {
    float2 f00 = bf2x(a00), f01 = bf2x(a01), f10 = bf2x(a10), f11 = bf2x(a11);
    float lo = w00*f00.x + w01*f01.x + w10*f10.x + w11*f11.x;
    float hi = w00*f00.y + w01*f01.y + w10*f10.y + w11*f11.y;
    return pack2bf(lo, hi);
}
__device__ __forceinline__ v8f wmma_bf16(const FragBF& a, const FragBF& b, v8f c) {
    return __builtin_amdgcn_wmma_f32_16x16x32_bf16(false, a.v, false, b.v,
                                                   (short)0, c, false, false);
}
// Load a lane-contiguous fragment (two b128 LDS reads)
__device__ __forceinline__ FragBF load_frag_lds(const uint32_t* p) {
    FragBF f;
    const uint4* q = (const uint4*)p;
    uint4 q0 = q[0], q1 = q[1];
    f.u[0]=q0.x; f.u[1]=q0.y; f.u[2]=q0.z; f.u[3]=q0.w;
    f.u[4]=q1.x; f.u[5]=q1.y; f.u[6]=q1.z; f.u[7]=q1.w;
    return f;
}

// ---------------- Kernel 0a: pre-swizzle weights into per-lane WMMA A-fragment order
// A2f layout: [m_tile(4)][chunk(18)][lane(32)][v(8)]  (u32, bf16 pair packed)
// A1f layout: [m_tile(2)][chunk(18)][lane(32)][v(8)]  rows >= 27 zero-padded
__global__ void pack_weights_kernel(const float* __restrict__ offw,
                                    const float* __restrict__ dcnw,
                                    uint32_t* __restrict__ A1f,
                                    uint32_t* __restrict__ A2f) {
    int idx = blockIdx.x * 256 + threadIdx.x;
    if (idx < 4*18*32*8) {
        int v     = idx & 7;
        int lane  = (idx >> 3) & 31;
        int chunk = (idx >> 8) % 18;
        int mt    = idx / 4608;
        int halfl = lane >> 4;
        int row   = mt*16 + (lane & 15);
        int kpair = chunk*16 + v + (halfl << 3);
        int k0    = kpair * 2;
        int tap   = k0 >> 6;      // k = tap*64 + c
        int c     = k0 & 63;      // even, c+1 same tap
        float a = dcnw[(row*CIN + c    )*9 + tap];
        float b = dcnw[(row*CIN + c + 1)*9 + tap];
        A2f[idx] = pack2bf(a, b);
    }
    int i1 = idx - 4*18*32*8;
    if (i1 >= 0 && i1 < 2*18*32*8) {
        int v     = i1 & 7;
        int lane  = (i1 >> 3) & 31;
        int chunk = (i1 >> 8) % 18;
        int mt    = i1 / 4608;
        int halfl = lane >> 4;
        int row   = mt*16 + (lane & 15);
        int kpair = chunk*16 + v + (halfl << 3);
        int k0    = kpair * 2;
        int tap   = k0 >> 6;
        int c     = k0 & 63;
        float a = 0.f, b = 0.f;
        if (row < 27) {
            a = offw[(row*CIN + c    )*9 + tap];
            b = offw[(row*CIN + c + 1)*9 + tap];
        }
        A1f[i1] = pack2bf(a, b);
    }
}

// ---------------- Kernel 0b: NCHW f32 -> NHWC bf16 (LDS-tiled transpose; coalesced both ways)
__global__ void __launch_bounds__(256)
transpose_nhwc_kernel(const float* __restrict__ x, uint16_t* __restrict__ xn) {
    __shared__ float tile[64][65];
    int wg   = blockIdx.x;          // BSZ*HD*2 = 2048
    int half = wg & 1;
    int y    = (wg >> 1) & (HD - 1);
    int b    = wg >> 8;
    int t    = threadIdx.x;
    int x0   = half * 64;
#pragma unroll
    for (int i = 0; i < 16; ++i) {
        int c  = i*4 + (t >> 6);
        int xx = t & 63;
        tile[c][xx] = x[((size_t)(b*CIN + c))*HW + y*WD + x0 + xx];
    }
    __syncthreads();
#pragma unroll
    for (int i = 0; i < 16; ++i) {
        int xx = i*4 + (t >> 6);
        int c  = t & 63;
        xn[((size_t)((b*HD + y)*WD) + x0 + xx)*CIN + c] = f2bf_bits(tile[c][xx]);
    }
}

// ---------------- Kernel 1: offset conv (M=32 padded, K=576, N tile = 64 px) via WMMA
// B kept in LDS in per-lane fragment order: Bf[buf][ntile(4)][lane(32)][v(8)]
__global__ void __launch_bounds__(256)
offset_conv_wmma(const uint16_t* __restrict__ xn, const uint32_t* __restrict__ A1f,
                 const float* __restrict__ offb, float* __restrict__ om) {
    __shared__ uint32_t Bf[2][4][32][8];
    int wg  = blockIdx.x;                // 2048
    int p0  = wg * 64;
    int b   = p0 / HW;
    int rem = p0 % HW;
    int h   = rem / WD;
    int w0  = rem % WD;
    int t     = threadIdx.x;
    int lane  = t & 31;
    int wid   = t >> 5;
    int m0    = (wid & 1) * 16;
    int ntile = wid >> 1;                // 0..3
    int halfl = lane >> 4;
    int l15   = lane & 15;
    const int pix = t & 63;
    const int grp = t >> 6;              // 0..3 -> 8-channel group
    const int lane_t = ((grp >= 2) ? 16 : 0) + (pix & 15);
    const int vb     = (grp & 1) * 4;
    const int ptile  = pix >> 4;

    auto build = [&](int chunk, int bi) {
        int kk = chunk >> 1;
        int c  = ((chunk & 1) << 5) + grp * 8;
        int yy = h + kk / 3 - 1;
        int xx = w0 + pix + kk % 3 - 1;
        uint4 ld = make_uint4(0u, 0u, 0u, 0u);
        if (yy >= 0 && yy < HD && xx >= 0 && xx < WD)
            ld = *(const uint4*)(xn + ((size_t)((b*HD + yy)*WD + xx))*CIN + c);
        *(uint4*)&Bf[bi][ptile][lane_t][vb] = ld;   // bf16 pairs already packed
    };

    v8f acc = {};
    build(0, 0);
    __syncthreads();
    for (int chunk = 0; chunk < NCHUNK; ++chunk) {
        int nxt = chunk + 1;
        if (nxt < NCHUNK) build(nxt, nxt & 1);

        const uint4* ap = (const uint4*)(A1f + (((size_t)(wid & 1)*NCHUNK + chunk)*32 + lane)*8);
        FragBF a;
        { uint4 a0 = ap[0], a1 = ap[1];
          a.u[0]=a0.x; a.u[1]=a0.y; a.u[2]=a0.z; a.u[3]=a0.w;
          a.u[4]=a1.x; a.u[5]=a1.y; a.u[6]=a1.z; a.u[7]=a1.w; }
        FragBF bm = load_frag_lds(&Bf[chunk & 1][ntile][lane][0]);
        acc = wmma_bf16(a, bm, acc);
        __syncthreads();
    }
#pragma unroll
    for (int v = 0; v < 8; ++v) {
        int m = m0 + v + (halfl << 3);
        if (m < 27)
            om[((size_t)(b*27 + m))*HW + h*WD + w0 + ntile*16 + l15] = acc[v] + offb[m];
    }
}

// ---------------- Kernel 2: fused deformable sampling + GEMM
// Tile: full image row (128 px) x Cout=64.  M=64, K=576, N=128.
// Each wave: 1 M-tile x 4 N-tiles -> 4 WMMAs per chunk from one A fragment.
// Sampling task = (pixel, 16-channel half); params computed once per pixel-half
// and prefetched one chunk ahead (double-buffered in registers).
// B in LDS in per-lane fragment order: Bf[buf][ntile(8)][lane(32)][v(8)] (16 KB).
__global__ void __launch_bounds__(256)
deform_gemm_wmma(const uint16_t* __restrict__ xn, const float* __restrict__ om,
                 const uint32_t* __restrict__ A2f, float* __restrict__ out) {
    __shared__ uint32_t Bf[2][8][32][8];
    int wg = blockIdx.x;              // BSZ*HD = 1024
    int h  = wg & (HD - 1);
    int b  = wg >> 7;
    int t     = threadIdx.x;
    int lane  = t & 31;
    int wid   = t >> 5;
    int m0    = (wid & 3) * 16;
    int q     = wid >> 2;             // wave handles ntiles q, q+2, q+4, q+6
    int halfl = lane >> 4;
    int l15   = lane & 15;
    const int pix    = t & 127;       // one pixel per thread
    const int hs     = t >> 7;        // 16-channel half of the 32-ch chunk
    const int ntile  = pix >> 4;
    const int lane_t = hs*16 + (pix & 15);
    const uint16_t* xb = xn + (size_t)b*HD*WD*CIN;
    const size_t om_row = (size_t)(b*27)*HW + h*WD;

    // double-buffered per-tap sampling state (corner offsets, weights, valid)
    uint32_t ca[2][4];
    float    cw[2][4];
    bool     cg[2][4];

    auto sample_params = [&](int kk, int s) {
        float dy = om[om_row + pix + (size_t)(2*kk    )*HW];
        float dx = om[om_row + pix + (size_t)(2*kk + 1)*HW];
        float mr = om[om_row + pix + (size_t)(18 + kk )*HW];
        float mask = 1.0f / (1.0f + __expf(-mr));
        float py = (float)(h   - 1 + kk / 3) + dy;
        float px = (float)(pix - 1 + kk % 3) + dx;
        float fy0 = floorf(py), fx0 = floorf(px);
        float fy = py - fy0, fx = px - fx0;
        int y0 = (int)fy0, x0 = (int)fx0;
        int y1 = y0 + 1,   x1 = x0 + 1;
        cw[s][0] = (1.f-fy)*(1.f-fx)*mask; cw[s][1] = (1.f-fy)*fx*mask;
        cw[s][2] = fy*(1.f-fx)*mask;       cw[s][3] = fy*fx*mask;
        bool yv0 = (y0 >= 0) & (y0 < HD), yv1 = (y1 >= 0) & (y1 < HD);
        bool xv0 = (x0 >= 0) & (x0 < WD), xv1 = (x1 >= 0) & (x1 < WD);
        cg[s][0] = yv0 & xv0; cg[s][1] = yv0 & xv1;
        cg[s][2] = yv1 & xv0; cg[s][3] = yv1 & xv1;
        ca[s][0] = (uint32_t)((y0*WD + x0)*CIN); ca[s][1] = (uint32_t)((y0*WD + x1)*CIN);
        ca[s][2] = (uint32_t)((y1*WD + x0)*CIN); ca[s][3] = (uint32_t)((y1*WD + x1)*CIN);
    };

    auto build = [&](int chunk, int bi, int s) {
        int c = ((chunk & 1) << 5) + hs * 16;    // this thread's 16 channels
        const uint16_t* xc = xb + c;
        uint4 z = make_uint4(0u,0u,0u,0u);
        uint4 l0a = z, l0b = z, l1a = z, l1b = z, l2a = z, l2b = z, l3a = z, l3b = z;
        if (cg[s][0]) { const uint4* p = (const uint4*)(xc + ca[s][0]); l0a = p[0]; l0b = p[1]; }
        if (cg[s][1]) { const uint4* p = (const uint4*)(xc + ca[s][1]); l1a = p[0]; l1b = p[1]; }
        if (cg[s][2]) { const uint4* p = (const uint4*)(xc + ca[s][2]); l2a = p[0]; l2b = p[1]; }
        if (cg[s][3]) { const uint4* p = (const uint4*)(xc + ca[s][3]); l3a = p[0]; l3b = p[1]; }
        float w0_ = cw[s][0], w1_ = cw[s][1], w2_ = cw[s][2], w3_ = cw[s][3];
        uint4 r0, r1;
        r0.x = lerp_pack(l0a.x, l1a.x, l2a.x, l3a.x, w0_, w1_, w2_, w3_);
        r0.y = lerp_pack(l0a.y, l1a.y, l2a.y, l3a.y, w0_, w1_, w2_, w3_);
        r0.z = lerp_pack(l0a.z, l1a.z, l2a.z, l3a.z, w0_, w1_, w2_, w3_);
        r0.w = lerp_pack(l0a.w, l1a.w, l2a.w, l3a.w, w0_, w1_, w2_, w3_);
        r1.x = lerp_pack(l0b.x, l1b.x, l2b.x, l3b.x, w0_, w1_, w2_, w3_);
        r1.y = lerp_pack(l0b.y, l1b.y, l2b.y, l3b.y, w0_, w1_, w2_, w3_);
        r1.z = lerp_pack(l0b.z, l1b.z, l2b.z, l3b.z, w0_, w1_, w2_, w3_);
        r1.w = lerp_pack(l0b.w, l1b.w, l2b.w, l3b.w, w0_, w1_, w2_, w3_);
        *(uint4*)&Bf[bi][ntile][lane_t][0] = r0;   // pair rows hs*8+0..3
        *(uint4*)&Bf[bi][ntile][lane_t][4] = r1;   // pair rows hs*8+4..7
    };

    v8f acc[4] = {v8f{}, v8f{}, v8f{}, v8f{}};
    int pc = 0;
    sample_params(0, 0);
    build(0, 0, 0);
    __syncthreads();
    for (int chunk = 0; chunk < NCHUNK; ++chunk) {
        int nxt = chunk + 1;
        if (nxt < NCHUNK) {
            if (nxt & 1) {                       // prefetch next tap's params
                int nt2 = (nxt >> 1) + 1;
                if (nt2 < 9) sample_params(nt2, pc ^ 1);
            } else {
                pc ^= 1;                         // switch to prefetched tap
            }
            build(nxt, nxt & 1, pc);
        }

        const uint4* ap = (const uint4*)(A2f + (((size_t)(wid & 3)*NCHUNK + chunk)*32 + lane)*8);
        FragBF a;
        { uint4 a0 = ap[0], a1 = ap[1];
          a.u[0]=a0.x; a.u[1]=a0.y; a.u[2]=a0.z; a.u[3]=a0.w;
          a.u[4]=a1.x; a.u[5]=a1.y; a.u[6]=a1.z; a.u[7]=a1.w; }
#pragma unroll
        for (int i = 0; i < 4; ++i) {
            FragBF bm = load_frag_lds(&Bf[chunk & 1][q + 2*i][lane][0]);
            acc[i] = wmma_bf16(a, bm, acc[i]);
        }
        __syncthreads();
    }
    size_t ob = (size_t)(b*COUT)*HW + h*WD;
#pragma unroll
    for (int i = 0; i < 4; ++i) {
        int nb = (q + 2*i) * 16;
#pragma unroll
        for (int v = 0; v < 8; ++v) {
            int m = m0 + v + (halfl << 3);
            out[ob + (size_t)m*HW + nb + l15] = acc[i][v];
        }
    }
}

// ---------------- Launch ----------------
extern "C" void kernel_launch(void* const* d_in, const int* in_sizes, int n_in,
                              void* d_out, int out_size, void* d_ws, size_t ws_size,
                              hipStream_t stream) {
    (void)in_sizes; (void)n_in; (void)out_size; (void)ws_size;
    const float* x    = (const float*)d_in[0];
    const float* offw = (const float*)d_in[1];
    const float* offb = (const float*)d_in[2];
    const float* dcnw = (const float*)d_in[3];
    float* out = (float*)d_out;

    char* ws = (char*)d_ws;
    uint16_t* xn  = (uint16_t*)(ws);                 // NHWC bf16: 16,777,216 B
    float*    om  = (float*)(ws + 16777216);         // 27-ch map f32: 14,155,776 B
    uint32_t* A1f = (uint32_t*)(ws + 30932992);      // 36,864 B
    uint32_t* A2f = (uint32_t*)(ws + 30969856);      // 73,728 B

    pack_weights_kernel<<<108, 256, 0, stream>>>(offw, dcnw, A1f, A2f);
    transpose_nhwc_kernel<<<BSZ*HD*2, 256, 0, stream>>>(x, xn);
    offset_conv_wmma<<<(BSZ*HW)/64, 256, 0, stream>>>(xn, A1f, offb, om);
    deform_gemm_wmma<<<BSZ*HD, 256, 0, stream>>>(xn, om, A2f, out);
}
